// CGNet_SSM_14396730376936
// MI455X (gfx1250) — matrix operand order repaired and checked by reading, not compiled, not run on hardware
//
#include <hip/hip_runtime.h>
#include <math.h>

// ---------------------------------------------------------------------------
// MI455X (gfx1250, wave32) implementation.
// Conv layers = implicit GEMM via v_wmma_f32_16x16x32_bf16.
//   - block tile 128(M) x 64(N), 8 waves, each wave = 32x32 output (4 WMMAs/K-step)
//   - double-buffered LDS pipeline: next A tile (global->LDS) and next B tile
//     (Tensor Data Mover, tensor_load_to_lds + s_wait_tensorcnt) are staged
//     while WMMAs run on the current buffers; one barrier per K-step.
//   - all H/W are powers of two -> shift/mask indexing (no int division).
// ---------------------------------------------------------------------------

typedef __bf16 bf16;
typedef __attribute__((ext_vector_type(16))) __bf16 v16bf;
typedef __attribute__((ext_vector_type(8)))  float  v8f;
typedef unsigned int u32x4 __attribute__((ext_vector_type(4)));
typedef int          i32x4 __attribute__((ext_vector_type(4)));
typedef int          i32x8 __attribute__((ext_vector_type(8)));

#define EPSV 1e-5f

#if defined(__has_builtin)
#if __has_builtin(__builtin_amdgcn_tensor_load_to_lds) && __has_builtin(__builtin_amdgcn_s_wait_tensorcnt)
#define HAVE_TDM 1
#endif
#endif
#ifndef HAVE_TDM
#define HAVE_TDM 0
#endif
#if __has_include(<hip/amd_detail/amd_gfx1250_TDM.h>)
#define TDM_6ARG 1
#else
#define TDM_6ARG 0
#endif

static __device__ __forceinline__ float b2f(bf16 x) { return (float)x; }
static __device__ __forceinline__ bf16  f2b(float x) { return (bf16)x; }

union FragU { uint4 u[2]; v16bf v; };

// ---------------------------------------------------------------------------
// WMMA implicit-GEMM conv (NHWC bf16 activations, weights [tap][cout][cin]).
// ---------------------------------------------------------------------------
__global__ __launch_bounds__(256)
void conv_wmma(const bf16* __restrict__ in, const bf16* __restrict__ wt,
               bf16* __restrict__ out,
               const float* __restrict__ gptr, const float* __restrict__ bptr,
               int H, int W, int lgW, int lgHW, int Cin, int Cout,
               int taps, int relu, float smul)
{
    __shared__ __align__(16) bf16 As[2][128][32];   // double-buffered M x K tile
    __shared__ __align__(16) bf16 Bs[2][64][32];    // double-buffered cout x K tile

    const int t    = threadIdx.x;
    const int wave = t >> 5;
    const int lane = t & 31;
    const int wm   = wave & 3;        // M-wave 0..3  (rows wm*32 .. wm*32+31)
    const int wn   = wave >> 2;       // N-wave 0..1  (couts wn*32 .. wn*32+31)
    const int hf   = lane >> 4;       // half-wave
    const int rr   = lane & 15;

    const int blockM    = blockIdx.x * 128;
    const int coutBlock = blockIdx.y * 64;
    const int kk  = (taps == 9) ? 3 : 1;
    const int pad = kk >> 1;

    // A cooperative staging: one 16-element (32B) chunk per thread
    const int ar = t >> 1;            // row 0..127
    const int ac = (t & 1) * 16;      // col 0 or 16
    const int m  = blockM + ar;
    const int n0 = m >> lgHW;                         // H*W is a power of two
    const int prem = m & ((1 << lgHW) - 1);
    const int oy = prem >> lgW;
    const int ox = prem & (W - 1);

#if !HAVE_TDM
    // B cooperative staging fallback: 8 elements (16B) per thread
    const int bco  = t >> 2;          // cout 0..63
    const int bseg = (t & 3) * 8;     // k segment
#endif

    v8f acc00 = {}, acc01 = {}, acc10 = {}, acc11 = {};

    const int kbA = hf * 8;           // A frag: lanes<16 K{0..7,16..23}; lanes>=16 K{8..15,24..31}
    const int kbB = hf * 8;           // B frag: lanes<16 K{0..7,16..23}; lanes>=16 K{8..15,24..31}

    auto stageA = [&](int buf, int ky, int kx, int kc) {
        const int iy = oy + ky - pad;
        const int ix = ox + kx - pad;
        const bool inb = (iy >= 0) && (iy < H) && (ix >= 0) && (ix < W);
        uint4 av0 = make_uint4(0u, 0u, 0u, 0u), av1 = av0;
        if (inb) {
            const bf16* src = in + (((size_t)(((n0 * H + iy) << lgW) + ix)) * Cin) + kc + ac;
            av0 = *reinterpret_cast<const uint4*>(src);
            av1 = *reinterpret_cast<const uint4*>(src + 8);
            __builtin_prefetch(src + Cin, 0, 1);   // next pixel row stays warm
        }
        *reinterpret_cast<uint4*>(&As[buf][ar][ac])     = av0;
        *reinterpret_cast<uint4*>(&As[buf][ar][ac + 8]) = av1;
    };

    auto issueB = [&](int buf, int tap, int kc) {
        const bf16* wtile = wt + ((size_t)(tap * Cout + coutBlock)) * Cin + kc;
#if HAVE_TDM
        if (wave == 0) {
            unsigned long long ga = (unsigned long long)(uintptr_t)wtile;
            u32x4 g0;
            g0[0] = 1u;                                   // count=1, user mode
            g0[1] = (unsigned)(uintptr_t)&Bs[buf][0][0];  // lds_addr
            g0[2] = (unsigned)ga;                         // global_addr[31:0]
            g0[3] = (unsigned)((ga >> 32) & 0x01FFFFFFu) | (2u << 30); // addr[56:32], type=2
            i32x8 g1;
            g1[0] = (1 << 16);                            // wg_mask=0, data_size=2B
            g1[1] = (Cin & 0xFFFF) << 16;                 // tensor_dim0[15:0]
            g1[2] = ((Cin >> 16) & 0xFFFF) |
                    (((taps * Cout) & 0xFFFF) << 16);     // dim0 hi | tensor_dim1 lo
            g1[3] = (((taps * Cout) >> 16) & 0xFFFF) |
                    (32 << 16);                           // dim1 hi | tile_dim0 = 32
            g1[4] = 64;                                   // tile_dim1 = 64, tile_dim2 = 0
            g1[5] = Cin;                                  // tensor_dim0_stride lo
            g1[6] = 0;
            g1[7] = 0;
            i32x4 g2 = {0, 0, 0, 0}, g3 = {0, 0, 0, 0};
#if TDM_6ARG
            i32x8 g4 = {0, 0, 0, 0, 0, 0, 0, 0};
            __builtin_amdgcn_tensor_load_to_lds(g0, g1, g2, g3, g4, 0);
#else
            __builtin_amdgcn_tensor_load_to_lds(g0, g1, g2, g3, 0);
#endif
        }
#else
        uint4 bv = *reinterpret_cast<const uint4*>(wtile + (size_t)bco * Cin + bseg);
        *reinterpret_cast<uint4*>(&Bs[buf][bco][bseg]) = bv;
#endif
    };

    auto waitB = [&]() {
#if HAVE_TDM
        if (wave == 0) __builtin_amdgcn_s_wait_tensorcnt(0);
#endif
    };

    // ---- pipelined K loop over (tap, kc) ----
    const int steps = taps * (Cin >> 5);
    int tap = 0, ky = 0, kx = 0, kc = 0;

    issueB(0, 0, 0);
    stageA(0, 0, 0, 0);
    waitB();
    __syncthreads();

    for (int s = 0; s < steps; ++s) {
        const int cb = s & 1;
        // advance (tap, kc) state without division
        int ntap = tap, nky = ky, nkx = kx, nkc = kc + 32;
        if (nkc == Cin) { nkc = 0; ++ntap; ++nkx; if (nkx == kk) { nkx = 0; ++nky; } }
        const bool more = (s + 1 < steps);

        if (more) {                       // stage next tiles into the other buffer
            issueB(cb ^ 1, ntap, nkc);
            stageA(cb ^ 1, nky, nkx, nkc);
        }

        // ---- fragments from current buffer (aligned 16B LDS loads) ----
        FragU a0, a1, b0, b1;
        const bf16* arow0 = &As[cb][wm * 32 + rr][0];
        const bf16* arow1 = &As[cb][wm * 32 + 16 + rr][0];
        a0.u[0] = *reinterpret_cast<const uint4*>(arow0 + kbA);
        a0.u[1] = *reinterpret_cast<const uint4*>(arow0 + kbA + 16);
        a1.u[0] = *reinterpret_cast<const uint4*>(arow1 + kbA);
        a1.u[1] = *reinterpret_cast<const uint4*>(arow1 + kbA + 16);
        const bf16* brow0 = &Bs[cb][wn * 32 + rr][0];
        const bf16* brow1 = &Bs[cb][wn * 32 + 16 + rr][0];
        b0.u[0] = *reinterpret_cast<const uint4*>(brow0 + kbB);
        b0.u[1] = *reinterpret_cast<const uint4*>(brow0 + kbB + 16);
        b1.u[0] = *reinterpret_cast<const uint4*>(brow1 + kbB);
        b1.u[1] = *reinterpret_cast<const uint4*>(brow1 + kbB + 16);

        acc00 = __builtin_amdgcn_wmma_f32_16x16x32_bf16(false, a0.v, false, b0.v, (short)0, acc00, false, false);
        acc01 = __builtin_amdgcn_wmma_f32_16x16x32_bf16(false, a0.v, false, b1.v, (short)0, acc01, false, false);
        acc10 = __builtin_amdgcn_wmma_f32_16x16x32_bf16(false, a1.v, false, b0.v, (short)0, acc10, false, false);
        acc11 = __builtin_amdgcn_wmma_f32_16x16x32_bf16(false, a1.v, false, b1.v, (short)0, acc11, false, false);

        if (more) waitB();                // TDM for next buffer must land before barrier
        __syncthreads();

        tap = ntap; ky = nky; kx = nkx; kc = nkc;
    }

    // ---- epilogue: fused BN scale/bias (+ReLU), bf16 NHWC stores ----
    const int nc0 = coutBlock + wn * 32 + rr;
    const int nc1 = nc0 + 16;
    float s0 = 1.f, s1 = 1.f, e0 = 0.f, e1 = 0.f;
    if (gptr) { s0 = gptr[nc0] * smul; s1 = gptr[nc1] * smul; }
    if (bptr) { e0 = bptr[nc0]; e1 = bptr[nc1]; }
#pragma unroll
    for (int v = 0; v < 8; ++v) {
        const int m0 = blockM + wm * 32 + v + hf * 8;        // subtile M rows
        const int m1 = m0 + 16;
        float v00 = acc00[v] * s0 + e0, v01 = acc01[v] * s1 + e1;
        float v10 = acc10[v] * s0 + e0, v11 = acc11[v] * s1 + e1;
        if (relu) {
            v00 = fmaxf(v00, 0.f); v01 = fmaxf(v01, 0.f);
            v10 = fmaxf(v10, 0.f); v11 = fmaxf(v11, 0.f);
        }
        out[(size_t)m0 * Cout + nc0] = f2b(v00);
        out[(size_t)m0 * Cout + nc1] = f2b(v01);
        out[(size_t)m1 * Cout + nc0] = f2b(v10);
        out[(size_t)m1 * Cout + nc1] = f2b(v11);
    }
}

// ---------------------------------------------------------------------------
// Weight packing: f32 OIHW -> bf16 [tap][cout][cin_pad] (zero-pads cin).
// ---------------------------------------------------------------------------
__global__ void pack_w(const float* __restrict__ w, bf16* __restrict__ wp,
                       int cout, int cin, int cinp, int kk)
{
    int i = blockIdx.x * 256 + threadIdx.x;
    int tot = kk * kk * cinp * cout;
    if (i >= tot) return;
    int ci = i % cinp;
    int p  = i / cinp;
    int co = p % cout;
    int tap = p / cout;
    int ky = tap / kk, kx = tap % kk;
    float v = (ci < cin) ? w[(((size_t)co * cin + ci) * kk + ky) * kk + kx] : 0.f;
    wp[i] = f2b(v);
}

// f32 NCHW (C=3) -> bf16 NHWC padded to 32 channels
__global__ void stem_pack(const float* __restrict__ in, bf16* __restrict__ out,
                          int N, int H, int W)
{
    int i = blockIdx.x * 256 + threadIdx.x;
    int tot = N * H * W * 32;
    if (i >= tot) return;
    int c = i & 31;
    int p = i >> 5;
    int x = p % W; p /= W;
    int y = p % H;
    int n = p / H;
    float v = 0.f;
    if (c < 3) v = in[(((size_t)n * 3 + c) * H + y) * W + x];
    out[i] = f2b(v);
}

// relu -> 2x2 maxpool (== max then relu), NHWC bf16; H,W are OUTPUT dims
__global__ void relupool(const bf16* __restrict__ in, bf16* __restrict__ out,
                         int N, int H, int W, int C)
{
    int i = blockIdx.x * 256 + threadIdx.x;
    int tot = N * H * W * C;
    if (i >= tot) return;
    int c = i % C;
    int p = i / C;
    int x = p % W; p /= W;
    int y = p % H;
    int n = p / H;
    size_t rs   = (size_t)2 * W * C;
    size_t base = ((size_t)(n * 2 * H + 2 * y) * (2 * W) + 2 * x) * C + c;
    float mA = fmaxf(b2f(in[base]),      b2f(in[base + C]));
    float mB = fmaxf(b2f(in[base + rs]), b2f(in[base + rs + C]));
    out[i] = f2b(fmaxf(fmaxf(mA, mB), 0.f));
}

// channel concat: out = [a | b] along C (NHWC)
__global__ void concat2(const bf16* __restrict__ a, const bf16* __restrict__ b,
                        bf16* __restrict__ out, int N, int H, int W, int Ca, int Cb)
{
    int Ct = Ca + Cb;
    int i = blockIdx.x * 256 + threadIdx.x;
    int tot = N * H * W * Ct;
    if (i >= tot) return;
    int c = i % Ct;
    int p = i / Ct;
    out[i] = (c < Ca) ? a[(size_t)p * Ca + c] : b[(size_t)p * Cb + (c - Ca)];
}

// align-corners bilinear resize, NHWC, T in {float, __bf16}
template <typename T>
__global__ void resize_ac_k(const T* __restrict__ in, T* __restrict__ out,
                            int N, int C, int H, int W, int OH, int OW)
{
    int i = blockIdx.x * 256 + threadIdx.x;
    int tot = N * OH * OW * C;
    if (i >= tot) return;
    int c = i % C;
    int p = i / C;
    int x = p % OW; p /= OW;
    int y = p % OH;
    int n = p / OH;
    float sy = (OH > 1) ? (float)y * (float)(H - 1) / (float)(OH - 1) : 0.f;
    float sx = (OW > 1) ? (float)x * (float)(W - 1) / (float)(OW - 1) : 0.f;
    int y0 = (int)floorf(sy); int y1 = min(y0 + 1, H - 1);
    int x0 = (int)floorf(sx); int x1 = min(x0 + 1, W - 1);
    float wy = sy - (float)y0, wx = sx - (float)x0;
    const T* bp = in + ((size_t)n * H) * W * C + c;
    float g00 = (float)bp[((size_t)y0 * W + x0) * C];
    float g01 = (float)bp[((size_t)y0 * W + x1) * C];
    float g10 = (float)bp[((size_t)y1 * W + x0) * C];
    float g11 = (float)bp[((size_t)y1 * W + x1) * C];
    float top = g00 * (1.f - wx) + g01 * wx;
    float bot = g10 * (1.f - wx) + g11 * wx;
    out[i] = (T)(top * (1.f - wy) + bot * wy);
}

// Fm = x + alpha * clip(prior, -1, 1)   (prior broadcast over channels)
__global__ void rpss_fm(const bf16* __restrict__ l, const float* __restrict__ prior,
                        const float* __restrict__ alpha, bf16* __restrict__ Fm,
                        int tot, int C)
{
    int i = blockIdx.x * 256 + threadIdx.x;
    if (i >= tot) return;
    int p = i / C;
    float pr = fminf(fmaxf(prior[p], -1.f), 1.f);
    Fm[i] = f2b(b2f(l[i]) + alpha[0] * pr);
}

// out = Fm + gamma * o
__global__ void rpss_res(const bf16* __restrict__ Fm, const bf16* __restrict__ o,
                         const float* __restrict__ gamma, bf16* __restrict__ out, int tot)
{
    int i = blockIdx.x * 256 + threadIdx.x;
    if (i >= tot) return;
    out[i] = f2b(b2f(Fm[i]) + gamma[0] * b2f(o[i]));
}

// SSM scan along W: one thread per (n,y,c); writes h into tmp (f32)
__global__ void scan_w(const bf16* __restrict__ xp, float* __restrict__ tmp,
                       const float* __restrict__ Ap, const float* __restrict__ Bp,
                       int N, int H, int W, int C)
{
    int id = blockIdx.x * 256 + threadIdx.x;
    int tot = N * H * C;
    if (id >= tot) return;
    int c = id % C;
    int ny = id / C;
    int y = ny % H;
    int n = ny / H;
    float A = tanhf(Ap[c]), B = Bp[c];
    float h = 0.f;
    size_t base = ((size_t)(n * H + y) * W) * C + c;
    for (int x = 0; x < W; ++x) {
        size_t o = base + (size_t)x * C;
        h = A * h + B * b2f(xp[o]);
        tmp[o] = h;
    }
}

// SSM scan along H: one thread per (n,x,c); adds h into tmp (f32)
__global__ void scan_h(const bf16* __restrict__ xp, float* __restrict__ tmp,
                       const float* __restrict__ Ap, const float* __restrict__ Bp,
                       int N, int H, int W, int C)
{
    int id = blockIdx.x * 256 + threadIdx.x;
    int tot = N * W * C;
    if (id >= tot) return;
    int c = id % C;
    int nx = id / C;
    int x = nx % W;
    int n = nx / W;
    float A = tanhf(Ap[c]), B = Bp[c];
    float h = 0.f;
    size_t stride = (size_t)W * C;
    size_t base = ((size_t)n * H) * stride + (size_t)x * C + c;
    for (int y = 0; y < H; ++y) {
        size_t o = base + (size_t)y * stride;
        h = A * h + B * b2f(xp[o]);
        tmp[o] += h;
    }
}

__global__ void cvt_f2b(const float* __restrict__ in, bf16* __restrict__ out, int tot)
{
    int i = blockIdx.x * 256 + threadIdx.x;
    if (i >= tot) return;
    out[i] = f2b(in[i]);
}

// tiny conv head (Cout=1), f32 weights OIHW, f32 output (n,y,x)
__global__ void conv_small(const bf16* __restrict__ in, const float* __restrict__ w,
                           const float* __restrict__ b, float* __restrict__ out,
                           int N, int H, int W, int Cin, int kk)
{
    int i = blockIdx.x * 256 + threadIdx.x;
    int tot = N * H * W;
    if (i >= tot) return;
    int x = i % W;
    int p = i / W;
    int y = p % H;
    int n = p / H;
    int pad = kk / 2;
    float acc = b[0];
    for (int ky = 0; ky < kk; ++ky) {
        int iy = y + ky - pad;
        if (iy < 0 || iy >= H) continue;
        for (int kx = 0; kx < kk; ++kx) {
            int ix = x + kx - pad;
            if (ix < 0 || ix >= W) continue;
            const bf16* src = in + (((size_t)(n * H + iy)) * W + ix) * Cin;
            for (int ci = 0; ci < Cin; ++ci)
                acc += b2f(src[ci]) * w[((size_t)ci * kk + ky) * kk + kx];
        }
    }
    out[i] = acc;
}

// ---------------------------------------------------------------------------
// Host orchestration
// ---------------------------------------------------------------------------
struct CBN { const float *b, *g, *w; };
struct RP  { const float *A, *Bm, *alpha, *bi, *bo, *gamma, *wi, *wo; };

static inline dim3 G1(size_t n) { return dim3((unsigned)((n + 255) / 256)); }
static inline int ilog2(int v) { int l = 0; while ((1 << l) < v) ++l; return l; }

extern "C" void kernel_launch(void* const* d_in, const int* in_sizes, int n_in,
                              void* d_out, int out_size, void* d_ws, size_t ws_size,
                              hipStream_t stream)
{
    (void)in_sizes; (void)n_in; (void)out_size; (void)ws_size;
    const int N = 2;
    const float smulBN = 1.0f / sqrtf(1.0f + EPSV);

    // ---- flattened pytree walk (dict keys sorted) ----
    int cur = 0;
    auto nxt = [&]() { return (const float*)d_in[cur++]; };
    auto getCBN = [&]() { CBN c; c.b = nxt(); c.g = nxt(); c.w = nxt(); return c; };
    auto getRP  = [&]() { RP r; r.A = nxt(); r.Bm = nxt(); r.alpha = nxt(); r.bi = nxt();
                          r.bo = nxt(); r.gamma = nxt(); r.wi = nxt(); r.wo = nxt(); return r; };

    const float* Aimg = nxt();                    // 'A'
    const float* Bimg = nxt();                    // 'B'
    CBN dec0 = getCBN();                          // dec0: b,g,w
    const float* dec1_b = nxt(); const float* dec1_w = nxt();   // dec1: b,w
    CBN df0 = getCBN();
    const float* df1_b = nxt(); const float* df1_w = nxt();
    CBN dm2 = getCBN(), dm3 = getCBN(), dm4 = getCBN();
    CBN d1[2]; for (int i = 0; i < 2; ++i) d1[i] = getCBN();
    CBN d2[3]; for (int i = 0; i < 3; ++i) d2[i] = getCBN();
    CBN d3[3]; for (int i = 0; i < 3; ++i) d3[i] = getCBN();
    CBN d4[3]; for (int i = 0; i < 3; ++i) d4[i] = getCBN();
    CBN inc[2]; for (int i = 0; i < 2; ++i) inc[i] = getCBN();
    CBN r1 = getCBN(), r2 = getCBN(), r3 = getCBN(), r4 = getCBN();
    RP rp1 = getRP(), rp2 = getRP(), rp3 = getRP();

    // ---- workspace bump allocator ----
    char* wsb = (char*)d_ws;
    size_t off = 0;
    auto alloc = [&](size_t bytes) -> void* {
        void* p = wsb + off;
        off = (off + bytes + 255) & ~(size_t)255;
        return p;
    };

    // ---- pack all conv weights to bf16 [tap][cout][cin_pad] ----
    auto packW = [&](const float* w, int cin, int cinp, int cout, int kk) -> const bf16* {
        size_t tot = (size_t)kk * kk * cinp * cout;
        bf16* wp = (bf16*)alloc(tot * 2);
        pack_w<<<G1(tot), 256, 0, stream>>>(w, wp, cout, cin, cinp, kk);
        return wp;
    };
    const bf16 *Winc0 = packW(inc[0].w, 3, 32, 64, 3);
    const bf16 *Winc1 = packW(inc[1].w, 64, 64, 64, 3);
    const bf16 *Wd1a = packW(d1[0].w, 64, 64, 128, 3),  *Wd1b = packW(d1[1].w, 128, 128, 128, 3);
    const bf16 *Wd2a = packW(d2[0].w, 128, 128, 256, 3), *Wd2b = packW(d2[1].w, 256, 256, 256, 3),
               *Wd2c = packW(d2[2].w, 256, 256, 256, 3);
    const bf16 *Wd3a = packW(d3[0].w, 256, 256, 512, 3), *Wd3b = packW(d3[1].w, 512, 512, 512, 3),
               *Wd3c = packW(d3[2].w, 512, 512, 512, 3);
    const bf16 *Wd4a = packW(d4[0].w, 512, 512, 512, 3), *Wd4b = packW(d4[1].w, 512, 512, 512, 3),
               *Wd4c = packW(d4[2].w, 512, 512, 512, 3);
    const bf16 *Wr1 = packW(r1.w, 256, 256, 128, 3),  *Wr2 = packW(r2.w, 512, 512, 256, 3);
    const bf16 *Wr3 = packW(r3.w, 1024, 1024, 512, 3), *Wr4 = packW(r4.w, 1024, 1024, 512, 3);
    const bf16 *Wdec0 = packW(dec0.w, 512, 512, 64, 3);
    const bf16 *Wdm4 = packW(dm4.w, 1024, 1024, 512, 3), *Wdm3 = packW(dm3.w, 768, 768, 256, 3),
               *Wdm2 = packW(dm2.w, 384, 384, 128, 3);
    const bf16 *Wdf0 = packW(df0.w, 128, 128, 64, 3);
    const bf16 *Wwi1 = packW(rp1.wi, 128, 128, 128, 1), *Wwo1 = packW(rp1.wo, 128, 128, 128, 1);
    const bf16 *Wwi2 = packW(rp2.wi, 256, 256, 128, 1), *Wwo2 = packW(rp2.wo, 128, 128, 256, 1);
    const bf16 *Wwi3 = packW(rp3.wi, 512, 512, 256, 1), *Wwo3 = packW(rp3.wo, 256, 256, 512, 1);

    // ---- conv launcher: block tile 128M x 64N ----
    auto conv = [&](const bf16* in, const bf16* wp, const float* g, const float* bia,
                    bf16* out, int H, int W, int Cin, int Cout, int kk, int relu, int bn) {
        int taps = kk * kk;
        int M = N * H * W;
        dim3 gr(M / 128, Cout / 64);
        conv_wmma<<<gr, 256, 0, stream>>>(in, wp, out, g, bia, H, W,
                                          ilog2(W), ilog2(H * W), Cin, Cout,
                                          taps, relu, bn ? smulBN : 1.0f);
    };

    // ---- backbone scratch (ping-pong) + persistent features ----
    bf16* s0 = (bf16*)alloc((size_t)N * 256 * 256 * 64 * 2);
    bf16* s1 = (bf16*)alloc((size_t)N * 256 * 256 * 64 * 2);
    bf16 *fa1 = (bf16*)alloc((size_t)N * 128 * 128 * 128 * 2),
         *fa2 = (bf16*)alloc((size_t)N * 64 * 64 * 256 * 2),
         *fa3 = (bf16*)alloc((size_t)N * 32 * 32 * 512 * 2),
         *fa4 = (bf16*)alloc((size_t)N * 16 * 16 * 512 * 2);
    bf16 *fb1 = (bf16*)alloc((size_t)N * 128 * 128 * 128 * 2),
         *fb2 = (bf16*)alloc((size_t)N * 64 * 64 * 256 * 2),
         *fb3 = (bf16*)alloc((size_t)N * 32 * 32 * 512 * 2),
         *fb4 = (bf16*)alloc((size_t)N * 16 * 16 * 512 * 2);

    auto backbone = [&](const float* img, bf16* L1, bf16* L2, bf16* L3, bf16* L4) {
        stem_pack<<<G1((size_t)N * 256 * 256 * 32), 256, 0, stream>>>(img, s0, N, 256, 256);
        conv(s0, Winc0, inc[0].g, inc[0].b, s1, 256, 256, 32, 64, 3, 1, 1);
        conv(s1, Winc1, inc[1].g, inc[1].b, s0, 256, 256, 64, 64, 3, 0, 1);
        // down1
        relupool<<<G1((size_t)N * 128 * 128 * 64), 256, 0, stream>>>(s0, s1, N, 128, 128, 64);
        conv(s1, Wd1a, d1[0].g, d1[0].b, s0, 128, 128, 64, 128, 3, 1, 1);
        conv(s0, Wd1b, d1[1].g, d1[1].b, L1, 128, 128, 128, 128, 3, 0, 1);
        // down2
        relupool<<<G1((size_t)N * 64 * 64 * 128), 256, 0, stream>>>(L1, s0, N, 64, 64, 128);
        conv(s0, Wd2a, d2[0].g, d2[0].b, s1, 64, 64, 128, 256, 3, 1, 1);
        conv(s1, Wd2b, d2[1].g, d2[1].b, s0, 64, 64, 256, 256, 3, 1, 1);
        conv(s0, Wd2c, d2[2].g, d2[2].b, L2, 64, 64, 256, 256, 3, 0, 1);
        // down3
        relupool<<<G1((size_t)N * 32 * 32 * 256), 256, 0, stream>>>(L2, s0, N, 32, 32, 256);
        conv(s0, Wd3a, d3[0].g, d3[0].b, s1, 32, 32, 256, 512, 3, 1, 1);
        conv(s1, Wd3b, d3[1].g, d3[1].b, s0, 32, 32, 512, 512, 3, 1, 1);
        conv(s0, Wd3c, d3[2].g, d3[2].b, L3, 32, 32, 512, 512, 3, 0, 1);
        // down4
        relupool<<<G1((size_t)N * 16 * 16 * 512), 256, 0, stream>>>(L3, s0, N, 16, 16, 512);
        conv(s0, Wd4a, d4[0].g, d4[0].b, s1, 16, 16, 512, 512, 3, 1, 1);
        conv(s1, Wd4b, d4[1].g, d4[1].b, s0, 16, 16, 512, 512, 3, 1, 1);
        conv(s0, Wd4c, d4[2].g, d4[2].b, L4, 16, 16, 512, 512, 3, 0, 1);
    };
    backbone(Aimg, fa1, fa2, fa3, fa4);
    backbone(Bimg, fb1, fb2, fb3, fb4);

    // ---- reduce stages: basic(concat([b, a])) ----
    bf16 *cat1 = (bf16*)alloc((size_t)N * 128 * 128 * 256 * 2),
         *cat2 = (bf16*)alloc((size_t)N * 64 * 64 * 512 * 2),
         *cat3 = (bf16*)alloc((size_t)N * 32 * 32 * 1024 * 2),
         *cat4 = (bf16*)alloc((size_t)N * 16 * 16 * 1024 * 2);
    bf16 *L1o = (bf16*)alloc((size_t)N * 128 * 128 * 128 * 2),
         *L2o = (bf16*)alloc((size_t)N * 64 * 64 * 256 * 2),
         *L3o = (bf16*)alloc((size_t)N * 32 * 32 * 512 * 2),
         *L4o = (bf16*)alloc((size_t)N * 16 * 16 * 512 * 2);
    concat2<<<G1((size_t)N * 128 * 128 * 256), 256, 0, stream>>>(fb1, fa1, cat1, N, 128, 128, 128, 128);
    concat2<<<G1((size_t)N * 64 * 64 * 512), 256, 0, stream>>>(fb2, fa2, cat2, N, 64, 64, 256, 256);
    concat2<<<G1((size_t)N * 32 * 32 * 1024), 256, 0, stream>>>(fb3, fa3, cat3, N, 32, 32, 512, 512);
    concat2<<<G1((size_t)N * 16 * 16 * 1024), 256, 0, stream>>>(fb4, fa4, cat4, N, 16, 16, 512, 512);
    conv(cat1, Wr1, r1.g, r1.b, L1o, 128, 128, 256, 128, 3, 1, 1);
    conv(cat2, Wr2, r2.g, r2.b, L2o, 64, 64, 512, 256, 3, 1, 1);
    conv(cat3, Wr3, r3.g, r3.b, L3o, 32, 32, 1024, 512, 3, 1, 1);
    conv(cat4, Wr4, r4.g, r4.b, L4o, 16, 16, 1024, 512, 3, 1, 1);

    // ---- change-map head: cm = conv(basic(l4, dec0), dec1) ----
    bf16* t4 = (bf16*)alloc((size_t)N * 16 * 16 * 64 * 2);
    conv(L4o, Wdec0, dec0.g, dec0.b, t4, 16, 16, 512, 64, 3, 1, 1);
    float* cm = (float*)alloc((size_t)N * 16 * 16 * 4);
    conv_small<<<G1((size_t)N * 16 * 16), 256, 0, stream>>>(t4, dec1_w, dec1_b, cm, N, 16, 16, 64, 3);

    // ---- RPSS ----
    auto rpssRun = [&](const bf16* l, int H, int W, int C, int hid, const RP& P,
                       const bf16* Wwi, const bf16* Wwo, bf16* oute) {
        size_t px = (size_t)N * H * W;
        float* prior = (float*)alloc(px * 4);
        resize_ac_k<float><<<G1(px), 256, 0, stream>>>(cm, prior, N, 1, 16, 16, H, W);
        bf16* Fm = (bf16*)alloc(px * C * 2);
        rpss_fm<<<G1(px * C), 256, 0, stream>>>(l, prior, P.alpha, Fm, (int)(px * C), C);
        bf16* xp = (bf16*)alloc(px * hid * 2);
        conv(Fm, Wwi, nullptr, P.bi, xp, H, W, C, hid, 1, 0, 0);
        float* hs = (float*)alloc(px * hid * 4);
        scan_w<<<G1((size_t)N * H * hid), 256, 0, stream>>>(xp, hs, P.A, P.Bm, N, H, W, hid);
        scan_h<<<G1((size_t)N * W * hid), 256, 0, stream>>>(xp, hs, P.A, P.Bm, N, H, W, hid);
        bf16* hb = (bf16*)alloc(px * hid * 2);
        cvt_f2b<<<G1(px * hid), 256, 0, stream>>>(hs, hb, (int)(px * hid));
        bf16* ro = (bf16*)alloc(px * C * 2);
        conv(hb, Wwo, nullptr, P.bo, ro, H, W, hid, C, 1, 0, 0);
        rpss_res<<<G1(px * C), 256, 0, stream>>>(Fm, ro, P.gamma, oute, (int)(px * C));
    };

    // ---- decoder chain ----
    bf16* l3e = (bf16*)alloc((size_t)N * 32 * 32 * 512 * 2);
    rpssRun(L3o, 32, 32, 512, 256, rp3, Wwi3, Wwo3, l3e);
    bf16* u4 = (bf16*)alloc((size_t)N * 32 * 32 * 512 * 2);
    resize_ac_k<bf16><<<G1((size_t)N * 32 * 32 * 512), 256, 0, stream>>>(L4o, u4, N, 512, 16, 16, 32, 32);
    bf16* c4 = (bf16*)alloc((size_t)N * 32 * 32 * 1024 * 2);
    concat2<<<G1((size_t)N * 32 * 32 * 1024), 256, 0, stream>>>(u4, l3e, c4, N, 32, 32, 512, 512);
    bf16* f4 = (bf16*)alloc((size_t)N * 32 * 32 * 512 * 2);
    conv(c4, Wdm4, dm4.g, dm4.b, f4, 32, 32, 1024, 512, 3, 1, 1);

    bf16* l2e = (bf16*)alloc((size_t)N * 64 * 64 * 256 * 2);
    rpssRun(L2o, 64, 64, 256, 128, rp2, Wwi2, Wwo2, l2e);
    bf16* u3 = (bf16*)alloc((size_t)N * 64 * 64 * 512 * 2);
    resize_ac_k<bf16><<<G1((size_t)N * 64 * 64 * 512), 256, 0, stream>>>(f4, u3, N, 512, 32, 32, 64, 64);
    bf16* c3 = (bf16*)alloc((size_t)N * 64 * 64 * 768 * 2);
    concat2<<<G1((size_t)N * 64 * 64 * 768), 256, 0, stream>>>(u3, l2e, c3, N, 64, 64, 512, 256);
    bf16* f3 = (bf16*)alloc((size_t)N * 64 * 64 * 256 * 2);
    conv(c3, Wdm3, dm3.g, dm3.b, f3, 64, 64, 768, 256, 3, 1, 1);

    bf16* l1e = (bf16*)alloc((size_t)N * 128 * 128 * 128 * 2);
    rpssRun(L1o, 128, 128, 128, 128, rp1, Wwi1, Wwo1, l1e);
    bf16* u2 = (bf16*)alloc((size_t)N * 128 * 128 * 256 * 2);
    resize_ac_k<bf16><<<G1((size_t)N * 128 * 128 * 256), 256, 0, stream>>>(f3, u2, N, 256, 64, 64, 128, 128);
    bf16* c2 = (bf16*)alloc((size_t)N * 128 * 128 * 384 * 2);
    concat2<<<G1((size_t)N * 128 * 128 * 384), 256, 0, stream>>>(u2, l1e, c2, N, 128, 128, 256, 128);
    bf16* l1f = (bf16*)alloc((size_t)N * 128 * 128 * 128 * 2);
    conv(c2, Wdm2, dm2.g, dm2.b, l1f, 128, 128, 384, 128, 3, 1, 1);

    // ---- final head ----
    bf16* t1 = (bf16*)alloc((size_t)N * 128 * 128 * 64 * 2);
    conv(l1f, Wdf0, df0.g, df0.b, t1, 128, 128, 128, 64, 3, 1, 1);
    float* fmm = (float*)alloc((size_t)N * 128 * 128 * 4);
    conv_small<<<G1((size_t)N * 128 * 128), 256, 0, stream>>>(t1, df1_w, df1_b, fmm, N, 128, 128, 64, 1);

    // ---- outputs: (change_map, final_map) each (2,1,256,256) f32, concatenated ----
    float* outp = (float*)d_out;
    resize_ac_k<float><<<G1((size_t)N * 256 * 256), 256, 0, stream>>>(cm, outp, N, 1, 16, 16, 256, 256);
    resize_ac_k<float><<<G1((size_t)N * 256 * 256), 256, 0, stream>>>(fmm, outp + (size_t)N * 256 * 256,
                                                                      N, 1, 128, 128, 256, 256);
}
